// VisualGraph_11991548691265
// MI455X (gfx1250) — matrix operand order repaired and checked by reading, not compiled, not run on hardware
//
#include <hip/hip_runtime.h>
#include <hip/hip_bf16.h>
#include <math.h>

// Problem constants (match reference)
#define BB 512
#define CC 80
#define FF 768
#define HH 256
#define EPSV 1e-5f
#define THRV 0.5f
#define SCALEV 0.0625f   // 1/sqrt(256)

typedef __attribute__((ext_vector_type(16))) __bf16 v16bf;
typedef __attribute__((ext_vector_type(8)))  __bf16 v8bf;
typedef __attribute__((ext_vector_type(8)))  float  v8f;

__device__ __forceinline__ unsigned short f32_to_bf16_bits(float f) {
    unsigned int u = __float_as_uint(f);
    u += 0x7fffu + ((u >> 16) & 1u);   // round-to-nearest-even
    return (unsigned short)(u >> 16);
}

// ---------------------------------------------------------------------------
// Elementwise fp32 -> bf16 convert (coalesced)
// ---------------------------------------------------------------------------
__global__ __launch_bounds__(256)
void cvt_f32_bf16_kernel(const float* __restrict__ in, unsigned short* __restrict__ out, long n) {
    long i = (long)blockIdx.x * 256 + threadIdx.x;
    if (i < n) out[i] = f32_to_bf16_bits(in[i]);
}

// ---------------------------------------------------------------------------
// fp32 [z][R][C] -> bf16 [z][C][R]  via 16x16 LDS tiles (coalesced both ways)
// R, C multiples of 16. grid.x = (R/16)*(C/16), grid.z = batch.
// ---------------------------------------------------------------------------
__global__ __launch_bounds__(256)
void transpose_cvt_kernel(const float* __restrict__ in, unsigned short* __restrict__ out,
                          int R, int C) {
    __shared__ float tile[16][17];        // +1 pad: kill LDS bank conflicts
    const long zoff = (long)blockIdx.z * R * C;
    const int tilesC = C >> 4;
    const int tr = (blockIdx.x / tilesC) << 4;    // row-block origin
    const int tc = (blockIdx.x % tilesC) << 4;    // col-block origin
    const int ty = threadIdx.x >> 4;
    const int tx = threadIdx.x & 15;
    tile[ty][tx] = in[zoff + (long)(tr + ty) * C + (tc + tx)];
    __syncthreads();
    out[zoff + (long)(tc + ty) * R + (tr + tx)] = f32_to_bf16_bits(tile[tx][ty]);
}

// ---------------------------------------------------------------------------
// Batched bf16 WMMA GEMM with wave-level macro-tiles:
//   Out[z] = A[z] (MxK, row-major) * B[z] (KxN, supplied transposed: BT [N][K]) + bias
// Each wave computes a (BM*16) x (BN*16) macro-tile: A frags shared across BN,
// B frags shared across BM. 8 waves / 256-thread block.
// outMode: 0 = fp32, 1 = bf16, 2 = bf16 transposed per-batch (for V^T)
// Requires (M/16)%BM==0, (N/16)%BN==0, K%32==0.
// ---------------------------------------------------------------------------
template<int BM, int BN>
__global__ __launch_bounds__(256)
void gemm_bf16_wmma_kernel(const unsigned short* __restrict__ A, int lda, long strideAz,
                           const unsigned short* __restrict__ BT, int ldb, long strideBz,
                           void* __restrict__ Out, int ldo, long strideOz,
                           const float* __restrict__ bias, long strideBiasz,
                           int M, int N, int K, int outMode,
                           int trRowsPerBatch, int trLdn, long trStrideB) {
    const int z = blockIdx.z;
    A  += (long)z * strideAz;
    BT += (long)z * strideBz;

    const int tilesN = (N >> 4) / BN;
    const int tilesM = (M >> 4) / BM;
    const int wave = threadIdx.x >> 5;
    const int lane = threadIdx.x & 31;
    const int t = blockIdx.x * 8 + wave;
    if (t >= tilesM * tilesN) return;          // wave-uniform: EXEC all-1s inside
    const int tm = t / tilesN;
    const int tn = t - tm * tilesN;

    const int hi = lane >> 4;                  // lane group (0/1)
    const int lo = lane & 15;

    // Per-lane base pointers (ISA 7.12.2 fragment layouts):
    //  A lane: row = lo, K chunks {8*hi+0..7} and {8*hi+16..23}  -> 2x b128
    //  B lane ([N][K] storage): col = lo, K chunks {8*hi+0..7} and {8*hi+16..23}  -> 2x b128
    const unsigned short* Arow[BM];
    const unsigned short* Bcol[BN];
#pragma unroll
    for (int i = 0; i < BM; ++i)
        Arow[i] = A + (long)(tm * BM * 16 + i * 16 + lo) * lda + 8 * hi;
#pragma unroll
    for (int j = 0; j < BN; ++j)
        Bcol[j] = BT + (long)(tn * BN * 16 + j * 16 + lo) * ldb + 8 * hi;

    v8f acc[BM][BN];
#pragma unroll
    for (int i = 0; i < BM; ++i)
#pragma unroll
        for (int j = 0; j < BN; ++j)
            acc[i][j] = (v8f){};

    for (int k0 = 0; k0 < K; k0 += 32) {
        v16bf a[BM], b[BN];
#pragma unroll
        for (int i = 0; i < BM; ++i) {
            v8bf a0 = *reinterpret_cast<const v8bf*>(Arow[i] + k0);
            v8bf a1 = *reinterpret_cast<const v8bf*>(Arow[i] + k0 + 16);
            a[i] = __builtin_shufflevector(a0, a1, 0,1,2,3,4,5,6,7,8,9,10,11,12,13,14,15);
        }
#pragma unroll
        for (int j = 0; j < BN; ++j) {
            v8bf b0 = *reinterpret_cast<const v8bf*>(Bcol[j] + k0);
            v8bf b1 = *reinterpret_cast<const v8bf*>(Bcol[j] + k0 + 16);
            b[j] = __builtin_shufflevector(b0, b1, 0,1,2,3,4,5,6,7,8,9,10,11,12,13,14,15);
        }
#pragma unroll
        for (int i = 0; i < BM; ++i)
#pragma unroll
            for (int j = 0; j < BN; ++j)
                acc[i][j] = __builtin_amdgcn_wmma_f32_16x16x32_bf16(
                                false, a[i], false, b[j],
                                (short)0, acc[i][j], false, false);
    }

#pragma unroll
    for (int j = 0; j < BN; ++j) {
        const int n = tn * BN * 16 + j * 16 + lo;
        const float bvv = bias ? bias[(long)z * strideBiasz + n] : 0.0f;
#pragma unroll
        for (int i = 0; i < BM; ++i) {
#pragma unroll
            for (int r = 0; r < 8; ++r) {
                const int m = tm * BM * 16 + i * 16 + r + 8 * hi;
                const float val = acc[i][j][r] + bvv;
                if (outMode == 0) {
                    ((float*)Out)[(long)z * strideOz + (long)m * ldo + n] = val;
                } else if (outMode == 1) {
                    ((unsigned short*)Out)[(long)z * strideOz + (long)m * ldo + n] =
                        f32_to_bf16_bits(val);
                } else {
                    const int bb = m / trRowsPerBatch;
                    const int mr = m - bb * trRowsPerBatch;
                    ((unsigned short*)Out)[(long)bb * trStrideB + (long)n * trLdn + mr] =
                        f32_to_bf16_bits(val);
                }
            }
        }
    }
}

// ---------------------------------------------------------------------------
// Row LayerNorm (H=256) + ReLU (+ optional residual add) -> bf16
// classMod==0: gamma/beta indexed [t]; else indexed [(row%classMod)*H + t]
// ---------------------------------------------------------------------------
__global__ __launch_bounds__(256)
void ln_relu_kernel(const float* __restrict__ Z, const float* __restrict__ gamma,
                    const float* __restrict__ beta, const float* __restrict__ addend,
                    unsigned short* __restrict__ outBf, int classMod) {
    const int row = blockIdx.x;
    const int tid = threadIdx.x;
    const float val = Z[(long)row * HH + tid];

    __shared__ float s1[256];
    __shared__ float s2[256];
    s1[tid] = val;
    s2[tid] = val * val;
    __syncthreads();
    for (int s = 128; s > 0; s >>= 1) {
        if (tid < s) { s1[tid] += s1[tid + s]; s2[tid] += s2[tid + s]; }
        __syncthreads();
    }
    const float mean = s1[0] * (1.0f / HH);
    const float var  = s2[0] * (1.0f / HH) - mean * mean;

    float g, bt;
    if (classMod) {
        const int c = row % classMod;
        g  = gamma[(long)c * HH + tid];
        bt = beta [(long)c * HH + tid];
    } else {
        g  = gamma[tid];
        bt = beta [tid];
    }
    float y = (val - mean) * rsqrtf(var + EPSV) * g + bt;
    y = fmaxf(y, 0.0f);
    if (addend) y += addend[(long)row * HH + tid];
    outBf[(long)row * HH + tid] = f32_to_bf16_bits(y);
}

// ---------------------------------------------------------------------------
// scale + hard threshold + label mask + softmax over d (80) -> bf16 padded 96
// ---------------------------------------------------------------------------
__global__ __launch_bounds__(128)
void mask_softmax_kernel(const float* __restrict__ scores, const int* __restrict__ labels,
                         unsigned short* __restrict__ attn) {
    const int r = blockIdx.x;          // (b, cRow) flattened, 0..B*C-1
    const int b = r / CC;
    const int t = threadIdx.x;

    __shared__ float red[128];
    float val = 0.0f;
    float mval = -3.402823466e38f;
    if (t < CC) {
        float v = scores[(long)r * CC + t] * SCALEV;
        v = (v > THRV) ? v : 0.0f;
        const float lm = (float)labels[(long)b * CC + t] * 0.8f + 0.2f;
        v *= lm;
        val = v;
        mval = v;
    }
    red[t] = mval;
    __syncthreads();
    for (int s = 64; s > 0; s >>= 1) {
        if (t < s) red[t] = fmaxf(red[t], red[t + s]);
        __syncthreads();
    }
    const float mx = red[0];
    __syncthreads();
    const float e = (t < CC) ? expf(val - mx) : 0.0f;
    red[t] = e;
    __syncthreads();
    for (int s = 64; s > 0; s >>= 1) {
        if (t < s) red[t] += red[t + s];
        __syncthreads();
    }
    const float inv = 1.0f / red[0];
    if (t < 96) attn[(long)r * 96 + t] = (t < CC) ? f32_to_bf16_bits(e * inv) : (unsigned short)0;
}

// ---------------------------------------------------------------------------
template<int BM, int BN>
static void launch_gemm(const unsigned short* A, int lda, long sAz,
                        const unsigned short* BT, int ldb, long sBz,
                        void* O, int ldo, long sOz,
                        const float* bias, long sBiasz,
                        int M, int N, int K, int mode, int nz,
                        int trRows, int trLdn, long trStrideB, hipStream_t stream) {
    const int waveTiles = ((M >> 4) / BM) * ((N >> 4) / BN);
    dim3 grid((waveTiles + 7) / 8, 1, nz);
    gemm_bf16_wmma_kernel<BM, BN><<<grid, 256, 0, stream>>>(
        A, lda, sAz, BT, ldb, sBz, O, ldo, sOz, bias, sBiasz,
        M, N, K, mode, trRows, trLdn, trStrideB);
}

extern "C" void kernel_launch(void* const* d_in, const int* in_sizes, int n_in,
                              void* d_out, int out_size, void* d_ws, size_t ws_size,
                              hipStream_t stream) {
    (void)in_sizes; (void)n_in; (void)out_size; (void)ws_size;

    const float* x   = (const float*)d_in[0];
    const float* vf  = (const float*)d_in[1];
    const int*   lab = (const int*)  d_in[2];
    const float* Wp  = (const float*)d_in[3];
    const float* bp  = (const float*)d_in[4];
    const float* gp  = (const float*)d_in[5];
    const float* bep = (const float*)d_in[6];
    const float* Wc  = (const float*)d_in[7];
    const float* bc  = (const float*)d_in[8];
    const float* gc  = (const float*)d_in[9];
    const float* bec = (const float*)d_in[10];
    const float* Wq  = (const float*)d_in[11];
    const float* bq  = (const float*)d_in[12];
    const float* Wk  = (const float*)d_in[13];
    const float* bk  = (const float*)d_in[14];
    const float* Wv  = (const float*)d_in[15];
    const float* bv  = (const float*)d_in[16];
    const float* Wo  = (const float*)d_in[17];
    const float* bo  = (const float*)d_in[18];
    float* out = (float*)d_out;

    // ---- workspace bump allocator (all sizes multiples of 256B) ----
    char* p = (char*)d_ws;
    auto alloc = [&](size_t bytes) -> char* {
        char* r = p;
        p += (bytes + 255) & ~(size_t)255;
        return r;
    };
    unsigned short* vf16   = (unsigned short*)alloc((size_t)BB * FF * 2);          // vf bf16
    unsigned short* WpT    = (unsigned short*)alloc((size_t)HH * FF * 2);          // [N=H][K=F]
    unsigned short* WcT    = (unsigned short*)alloc((size_t)CC * HH * HH * 2);     // per-class [N][K]
    unsigned short* WqT    = (unsigned short*)alloc((size_t)HH * HH * 2);
    unsigned short* WkT    = (unsigned short*)alloc((size_t)HH * HH * 2);
    unsigned short* WvT    = (unsigned short*)alloc((size_t)HH * HH * 2);
    unsigned short* WoT    = (unsigned short*)alloc((size_t)HH * HH * 2);
    float*          z1     = (float*)         alloc((size_t)BB * HH * 4);          // vf@Wp + bp
    unsigned short* vis16  = (unsigned short*)alloc((size_t)BB * HH * 2);
    float*          zbuf   = (float*)         alloc((size_t)BB * CC * HH * 4);     // ensemble out
    unsigned short* xa16   = (unsigned short*)alloc((size_t)BB * CC * HH * 2);     // x + class_feats
    unsigned short* vT16   = (unsigned short*)alloc((size_t)BB * HH * 96 * 2);     // V^T padded K=96
    float*          sc32   = (float*)         alloc((size_t)BB * CC * CC * 4);     // raw scores
    unsigned short* attn16 = (unsigned short*)alloc((size_t)BB * CC * 96 * 2);     // softmax, padded
    unsigned short* out116 = (unsigned short*)alloc((size_t)BB * CC * HH * 2);     // attn@V
    // recycle zbuf (dead after LN2) for Q and K bf16
    unsigned short* q16 = (unsigned short*)zbuf;
    unsigned short* k16 = (unsigned short*)((char*)zbuf + (size_t)BB * CC * HH * 2);

    // ---- 1) convert visual features + pre-transpose all weights to bf16 ----
    {
        long n = (long)BB * FF;
        cvt_f32_bf16_kernel<<<dim3((unsigned)((n + 255) / 256)), 256, 0, stream>>>(vf, vf16, n);
    }
    transpose_cvt_kernel<<<dim3((FF >> 4) * (HH >> 4), 1, 1),  256, 0, stream>>>(Wp, WpT, FF, HH);
    transpose_cvt_kernel<<<dim3((HH >> 4) * (HH >> 4), 1, CC), 256, 0, stream>>>(Wc, WcT, HH, HH);
    transpose_cvt_kernel<<<dim3((HH >> 4) * (HH >> 4), 1, 1),  256, 0, stream>>>(Wq, WqT, HH, HH);
    transpose_cvt_kernel<<<dim3((HH >> 4) * (HH >> 4), 1, 1),  256, 0, stream>>>(Wk, WkT, HH, HH);
    transpose_cvt_kernel<<<dim3((HH >> 4) * (HH >> 4), 1, 1),  256, 0, stream>>>(Wv, WvT, HH, HH);
    transpose_cvt_kernel<<<dim3((HH >> 4) * (HH >> 4), 1, 1),  256, 0, stream>>>(Wo, WoT, HH, HH);

    // ---- 2) vis = relu(LN(vf@Wp + bp)) ----
    launch_gemm<2,4>(vf16, FF, 0, WpT, FF, 0, z1, HH, 0, bp, 0,
                     BB, HH, FF, /*f32*/0, 1, 1, 1, 0, stream);
    ln_relu_kernel<<<dim3(BB), 256, 0, stream>>>(z1, gp, bep, nullptr, vis16, 0);

    // ---- 3) z[b,c,:] = vis[b]@Wc[c] + bc[c]; xa = x + relu(LN(z)) ----
    launch_gemm<2,4>(vis16, HH, 0, WcT, HH, (long)HH * HH, zbuf, CC * HH, HH,
                     bc, HH, BB, HH, HH, /*f32*/0, CC, 1, 1, 0, stream);
    ln_relu_kernel<<<dim3(BB * CC), 256, 0, stream>>>(zbuf, gc, bec, x, xa16, CC);

    // ---- 4) Q, K, V projections (V stored transposed, K-padded to 96) ----
    launch_gemm<2,4>(xa16, HH, 0, WqT, HH, 0, q16, HH, 0, bq, 0,
                     BB * CC, HH, HH, /*bf16*/1, 1, 1, 1, 0, stream);
    launch_gemm<2,4>(xa16, HH, 0, WkT, HH, 0, k16, HH, 0, bk, 0,
                     BB * CC, HH, HH, /*bf16*/1, 1, 1, 1, 0, stream);
    launch_gemm<2,4>(xa16, HH, 0, WvT, HH, 0, vT16, 0, 0, bv, 0,
                     BB * CC, HH, HH, /*bf16 tr*/2, 1, CC, 96, (long)HH * 96, stream);

    // ---- 5) scores[b] = Q[b] K[b]^T  (raw; scale/mask in softmax kernel) ----
    launch_gemm<5,1>(q16, HH, (long)CC * HH, k16, HH, (long)CC * HH,
                     sc32, CC, (long)CC * CC, nullptr, 0,
                     CC, CC, HH, /*f32*/0, BB, 1, 1, 0, stream);

    // ---- 6) mask + softmax -> bf16 attn, zero-padded to K=96 ----
    mask_softmax_kernel<<<dim3(BB * CC), 128, 0, stream>>>(sc32, lab, attn16);

    // ---- 7) out1[b] = attn[b] @ V[b]  (K=96, pads contribute 0) ----
    launch_gemm<5,2>(attn16, 96, (long)CC * 96, vT16, 96, (long)HH * 96,
                     out116, HH, (long)CC * HH, nullptr, 0,
                     CC, HH, 96, /*bf16*/1, BB, 1, 1, 0, stream);

    // ---- 8) final: out = out1 @ Wo + bo (fp32 to d_out) ----
    launch_gemm<2,4>(out116, HH, 0, WoT, HH, 0, out, HH, 0, bo, 0,
                     BB * CC, HH, HH, /*f32*/0, 1, 1, 1, 0, stream);
}
